// RNN_86955907875240
// MI455X (gfx1250) — matrix operand order repaired and checked by reading, not compiled, not run on hardware
//
#include <hip/hip_runtime.h>
#include <hip/hip_bf16.h>

// ---------------- CDNA5 WMMA types ----------------
typedef __attribute__((ext_vector_type(16))) _Float16 v16h;
typedef __attribute__((ext_vector_type(8)))  _Float16 v8h;
typedef __attribute__((ext_vector_type(8)))  float    v8f;

// Problem constants (from reference)
#define TT 256
#define BB 64
#define EE 512
#define HH 512
#define KTAG 32
#define START_TAG 30
#define END_TAG 31
#define NEGV (-100000.0f)

__device__ inline float sigf(float x) { return 1.0f / (1.0f + __expf(-x)); }

__device__ inline v16h pack16(v8h lo, v8h hi) {
    v16h a;
#pragma unroll
    for (int i = 0; i < 8; ++i) { a[i] = lo[i]; a[i + 8] = hi[i]; }
    return a;
}

// ---------------- weight packing: [dir][4H][Kh+Kx] f16, K = [Whh | Wih] ----------------
__global__ void pack_cat_kernel(const float* __restrict__ whh,
                                const float* __restrict__ wih,
                                _Float16* __restrict__ dst, int Kx) {
    const int KK = 512 + Kx;
    const long long total = 2LL * 2048 * KK;
    for (long long i = (long long)blockIdx.x * blockDim.x + threadIdx.x; i < total;
         i += (long long)gridDim.x * blockDim.x) {
        int k = (int)(i % KK);
        long long gd = i / KK;
        int g = (int)(gd % 2048);
        int dir = (int)(gd / 2048);
        float v = (k < 512) ? whh[((size_t)dir * 2048 + g) * 512 + k]
                            : wih[((size_t)dir * 2048 + g) * Kx + (k - 512)];
        dst[i] = (_Float16)v;
    }
}

__global__ void cvt_half_kernel(const float* __restrict__ src, _Float16* __restrict__ dst, int n) {
    for (int i = blockIdx.x * blockDim.x + threadIdx.x; i < n; i += gridDim.x * blockDim.x)
        dst[i] = (_Float16)src[i];
}

// ---------------- embedding gather -> f16 ----------------
__global__ void embed_gather_kernel(const int* __restrict__ tokens,
                                    const float* __restrict__ embed,
                                    _Float16* __restrict__ x) {
    int tb = blockIdx.x;                 // t*B + b
    int tok = tokens[tb];
    const float* src = embed + (size_t)tok * EE;
    _Float16* dst = x + (size_t)tb * EE;
    for (int e = threadIdx.x; e < EE; e += blockDim.x) dst[e] = (_Float16)src[e];
}

// ---------------- state zero ----------------
__global__ void zero_state_kernel(_Float16* __restrict__ h, float* __restrict__ c) {
    int i = blockIdx.x * blockDim.x + threadIdx.x;
    if (i < 2 * 2 * BB * HH) h[i] = (_Float16)0.0f;   // 2 phases x 2 dirs
    if (i < 2 * BB * HH)     c[i] = 0.0f;             // 2 dirs
}

// ---------------- fused LSTM step: gates = [h|x] @ [Whh|Wih]^T + b, then cell update --------
// grid: (8 j-blocks, 2 dirs), block: 256 threads (8 waves)
__global__ __launch_bounds__(256)
void lstm_step_kernel(const _Float16* __restrict__ W,     // [2][2048][KK] f16
                      const float* __restrict__ bias,     // [2][2048]
                      const _Float16* __restrict__ xin,   // [T][B][XW] f16
                      _Float16* __restrict__ yout,        // [T][B][2H] f16
                      const _Float16* __restrict__ hin,   // [2][B][H] f16
                      _Float16* __restrict__ hout,        // [2][B][H] f16
                      float* __restrict__ cstate,         // [2][B][H]
                      int tstep, int KK, int XW) {
    const int dir = blockIdx.y;
    const int tt = (dir == 0) ? tstep : (TT - 1 - tstep);
    const int j0 = blockIdx.x * 64;
    const int wave = threadIdx.x >> 5;
    const int lane = threadIdx.x & 31;
    const int r = wave >> 1;        // row tile 0..3
    const int s = wave & 1;         // 32-col half of the j-block

    __shared__ _Float16 As[64 * 128];   // A chunk: 64 rows x 128 K-halves (16 KB)

    const _Float16* hprev = hin + (size_t)dir * BB * HH;
    _Float16* hnext = hout + (size_t)dir * BB * HH;
    float* cst = cstate + (size_t)dir * BB * HH;
    const _Float16* Wd = W + (size_t)dir * 2048 * KK;

    const v8f vzero = {0.f, 0.f, 0.f, 0.f, 0.f, 0.f, 0.f, 0.f};
    v8f acc[4][2];
#pragma unroll
    for (int g = 0; g < 4; ++g)
#pragma unroll
        for (int ct = 0; ct < 2; ++ct) acc[g][ct] = vzero;

    for (int k0 = 0; k0 < KK; k0 += 128) {
        // cooperative A-chunk stage: [h_prev | x_t] columns k0..k0+127
        {
            int b = threadIdx.x >> 2;
            int seg = threadIdx.x & 3;
            const _Float16* src = (k0 < 512)
                ? (hprev + (size_t)b * HH + k0 + seg * 32)
                : (xin + ((size_t)tt * BB + b) * XW + (k0 - 512) + seg * 32);
            v8h* d = (v8h*)&As[b * 128 + seg * 32];
            const v8h* sp = (const v8h*)src;
            d[0] = sp[0]; d[1] = sp[1]; d[2] = sp[2]; d[3] = sp[3];
        }
        __syncthreads();

        // prefetch next chunk's weights for this wave (L2 hint)
        if (k0 + 128 < KK) {
            int gc0 = j0 + s * 32 + (lane & 15);
            __builtin_prefetch(Wd + (size_t)gc0 * KK + (k0 + 128), 0, 1);
        }

#pragma unroll
        for (int q = 0; q < 4; ++q) {
            const int kq = q * 32;
            // A fragment (16x32 f16): lane<16 -> K {0..15}; lane>=16 -> {16..31}
            const _Float16* pa = &As[(r * 16 + (lane & 15)) * 128 + kq + ((lane >> 4) << 4)];
            v8h lo = *(const v8h*)pa;
            v8h hi = *(const v8h*)(pa + 8);
            v16h a = pack16(lo, hi);
#pragma unroll
            for (int g = 0; g < 4; ++g) {
#pragma unroll
                for (int ct = 0; ct < 2; ++ct) {
                    const int gcol = g * 512 + j0 + s * 32 + ct * 16 + (lane & 15);
                    const _Float16* pb =
                        Wd + (size_t)gcol * KK + (k0 + kq + ((lane >> 4) << 4));
                    v16h bfr = *(const v16h*)pb;  // 16 contiguous K-halves (32B aligned)
                    acc[g][ct] = __builtin_amdgcn_wmma_f32_16x16x32_f16(
                        false, a, false, bfr, (short)0, acc[g][ct], false, false);
                }
            }
        }
        __syncthreads();
    }

    // epilogue: gate nonlinearities + cell update (i/f/g/o co-resident per lane/vgpr slot)
#pragma unroll
    for (int ct = 0; ct < 2; ++ct) {
        const int j = j0 + s * 32 + ct * 16 + (lane & 15);
        const float bi = bias[dir * 2048 + 0 * 512 + j];
        const float bf = bias[dir * 2048 + 1 * 512 + j];
        const float bg = bias[dir * 2048 + 2 * 512 + j];
        const float bo = bias[dir * 2048 + 3 * 512 + j];
#pragma unroll
        for (int v = 0; v < 8; ++v) {
            const int b = r * 16 + ((lane >> 4) << 3) + v;
            float gi = acc[0][ct][v] + bi;
            float gf = acc[1][ct][v] + bf;
            float gg = acc[2][ct][v] + bg;
            float go = acc[3][ct][v] + bo;
            float cp = cst[(size_t)b * HH + j];
            float cn = sigf(gf) * cp + sigf(gi) * tanhf(gg);
            float hn = sigf(go) * tanhf(cn);
            cst[(size_t)b * HH + j] = cn;
            _Float16 hh = (_Float16)hn;
            hnext[(size_t)b * HH + j] = hh;
            yout[((size_t)tt * BB + b) * (2 * HH) + dir * HH + j] = hh;
        }
    }
}

// ---------------- feats = y1 @ lin_w^T + lin_b : M=16384, N=32, K=1024 (WMMA) -------------
__global__ __launch_bounds__(128)
void feats_kernel(const _Float16* __restrict__ y1,    // [T*B][1024]
                  const _Float16* __restrict__ lwh,   // [32][1024]
                  const float* __restrict__ lb,       // [32]
                  float* __restrict__ feats) {        // [T*B][32]
    const int wave = threadIdx.x >> 5;
    const int lane = threadIdx.x & 31;
    const int row0 = blockIdx.x * 64 + wave * 16;

    const v8f vzero = {0.f, 0.f, 0.f, 0.f, 0.f, 0.f, 0.f, 0.f};
    v8f acc[2] = {vzero, vzero};

    for (int k0 = 0; k0 < 1024; k0 += 32) {
        const _Float16* pa =
            y1 + (size_t)(row0 + (lane & 15)) * 1024 + k0 + ((lane >> 4) << 4);
        v8h lo = *(const v8h*)pa;
        v8h hi = *(const v8h*)(pa + 8);
        v16h a = pack16(lo, hi);
#pragma unroll
        for (int ct = 0; ct < 2; ++ct) {
            const _Float16* pb =
                lwh + (size_t)(ct * 16 + (lane & 15)) * 1024 + k0 + ((lane >> 4) << 4);
            v16h bfr = *(const v16h*)pb;
            acc[ct] = __builtin_amdgcn_wmma_f32_16x16x32_f16(
                false, a, false, bfr, (short)0, acc[ct], false, false);
        }
    }
#pragma unroll
    for (int ct = 0; ct < 2; ++ct) {
#pragma unroll
        for (int v = 0; v < 8; ++v) {
            int rr = row0 + ((lane >> 4) << 3) + v;
            int n = ct * 16 + (lane & 15);
            feats[(size_t)rr * KTAG + n] = acc[ct][v] + lb[n];
        }
    }
}

// ---------------- CRF: one wave (32 lanes = 32 tags) per batch element -------------------
__global__ __launch_bounds__(32)
void crf_kernel(const float* __restrict__ feats, const int* __restrict__ tokens,
                const int* __restrict__ tags, const float* __restrict__ trans,
                const int* __restrict__ lengths, float* __restrict__ loss_b) {
    const int b = blockIdx.x;
    const int k = threadIdx.x;
    __shared__ float sc[KTAG];
    __shared__ float tr[KTAG * KTAG];
    for (int i = k; i < KTAG * KTAG; i += 32) tr[i] = trans[i];
    sc[k] = (k == START_TAG) ? 0.0f : NEGV;
    __syncthreads();

    for (int t = 0; t < TT; ++t) {
        float emit = feats[((size_t)t * BB + b) * KTAG + k];
        int m = tokens[t * BB + b] > 0;
        float mx = -3.4e38f;
#pragma unroll 4
        for (int j = 0; j < KTAG; ++j) mx = fmaxf(mx, sc[j] + tr[k * KTAG + j]);
        float ssum = 0.0f;
#pragma unroll 4
        for (int j = 0; j < KTAG; ++j) ssum += __expf(sc[j] + tr[k * KTAG + j] - mx);
        float nv = mx + __logf(ssum) + emit;
        __syncthreads();
        if (m) sc[k] = nv;
        __syncthreads();
    }

    // log_z = logsumexp_k(sc[k] + trans[END][k])
    float v = sc[k] + tr[END_TAG * KTAG + k];
    float mx = v;
    for (int off = 16; off > 0; off >>= 1) mx = fmaxf(mx, __shfl_xor(mx, off, 32));
    float e = __expf(v - mx);
    for (int off = 16; off > 0; off >>= 1) e += __shfl_xor(e, off, 32);
    float logz = mx + __logf(e);

    // gold score, lane-parallel over time
    float g = 0.0f;
    for (int t = k; t < TT; t += 32) {
        int cur = tags[t * BB + b];
        int prev = (t == 0) ? START_TAG : tags[(t - 1) * BB + b];
        float m = (tokens[t * BB + b] > 0) ? 1.0f : 0.0f;
        g += m * (tr[cur * KTAG + prev] + feats[((size_t)t * BB + b) * KTAG + cur]);
    }
    for (int off = 16; off > 0; off >>= 1) g += __shfl_xor(g, off, 32);

    if (k == 0) {
        int last = tags[(TT - 1) * BB + b];
        float gold = g + tr[END_TAG * KTAG + last];
        loss_b[b] = (logz - gold) / (float)lengths[b];
    }
}

__global__ void reduce_kernel(const float* __restrict__ loss_b, float* __restrict__ out) {
    __shared__ float s[64];
    int k = threadIdx.x;
    s[k] = loss_b[k];
    __syncthreads();
    for (int off = 32; off > 0; off >>= 1) {
        if (k < off) s[k] += s[k + off];
        __syncthreads();
    }
    if (k == 0) out[0] = s[0];
}

// ---------------- host orchestration ----------------
extern "C" void kernel_launch(void* const* d_in, const int* in_sizes, int n_in,
                              void* d_out, int out_size, void* d_ws, size_t ws_size,
                              hipStream_t stream) {
    const int* tokens  = (const int*)d_in[0];
    const int* tags    = (const int*)d_in[1];
    const int* lengths = (const int*)d_in[2];
    const float* embed = (const float*)d_in[3];
    const float* wih0  = (const float*)d_in[4];
    const float* whh0  = (const float*)d_in[5];
    const float* b0    = (const float*)d_in[6];
    const float* wih1  = (const float*)d_in[7];
    const float* whh1  = (const float*)d_in[8];
    const float* b1    = (const float*)d_in[9];
    const float* lin_w = (const float*)d_in[10];
    const float* lin_b = (const float*)d_in[11];
    const float* trans = (const float*)d_in[12];

    char* p = (char*)d_ws;
    auto carve = [&](size_t bytes) -> char* {
        char* r = p;
        p += (bytes + 255) & ~(size_t)255;
        return r;
    };
    _Float16* W0  = (_Float16*)carve(2ull * 2048 * 1024 * 2);   // 8 MB
    _Float16* W1  = (_Float16*)carve(2ull * 2048 * 1536 * 2);   // 12 MB
    _Float16* LWH = (_Float16*)carve((size_t)KTAG * 1024 * 2);
    _Float16* X   = (_Float16*)carve((size_t)TT * BB * EE * 2); // 16 MB
    _Float16* Y0  = (_Float16*)carve((size_t)TT * BB * 2 * HH * 2); // 32 MB
    _Float16* Y1  = (_Float16*)carve((size_t)TT * BB * 2 * HH * 2); // 32 MB
    _Float16* HST = (_Float16*)carve(2ull * 2 * BB * HH * 2);   // 2 phases x 2 dirs
    float*    CST = (float*)carve(2ull * BB * HH * 4);
    float*    FE  = (float*)carve((size_t)TT * BB * KTAG * 4);  // 2 MB
    float*    LB  = (float*)carve(BB * 4);

    const size_t hphase = (size_t)2 * BB * HH;  // halves per phase (2 dirs)

    // weight prep + embedding
    pack_cat_kernel<<<2048, 256, 0, stream>>>(whh0, wih0, W0, 512);
    pack_cat_kernel<<<2048, 256, 0, stream>>>(whh1, wih1, W1, 1024);
    cvt_half_kernel<<<128, 256, 0, stream>>>(lin_w, LWH, KTAG * 1024);
    embed_gather_kernel<<<TT * BB, 128, 0, stream>>>(tokens, embed, X);

    // BiLSTM layer 0 (fused input+recurrent GEMM per step, fwd+bwd in one launch)
    zero_state_kernel<<<512, 256, 0, stream>>>(HST, CST);
    for (int t = 0; t < TT; ++t) {
        const _Float16* hin = HST + (size_t)(t & 1) * hphase;
        _Float16* hout = HST + (size_t)((t + 1) & 1) * hphase;
        lstm_step_kernel<<<dim3(8, 2), 256, 0, stream>>>(W0, b0, X, Y0, hin, hout, CST,
                                                         t, 1024, 512);
    }

    // BiLSTM layer 1
    zero_state_kernel<<<512, 256, 0, stream>>>(HST, CST);
    for (int t = 0; t < TT; ++t) {
        const _Float16* hin = HST + (size_t)(t & 1) * hphase;
        _Float16* hout = HST + (size_t)((t + 1) & 1) * hphase;
        lstm_step_kernel<<<dim3(8, 2), 256, 0, stream>>>(W1, b1, Y0, Y1, hin, hout, CST,
                                                         t, 1536, 1024);
    }

    // tag logits + CRF
    feats_kernel<<<TT * BB / 64, 128, 0, stream>>>(Y1, LWH, lin_b, FE);
    crf_kernel<<<BB, 32, 0, stream>>>(FE, tokens, tags, trans, lengths, LB);
    reduce_kernel<<<1, 64, 0, stream>>>(LB, (float*)d_out);
}
